// MY_expGCN_parallel_model_26207890440883
// MI455X (gfx1250) — matrix operand order repaired and checked by reading, not compiled, not run on hardware
//
#include <hip/hip_runtime.h>
#include <math.h>

// ---------------------------------------------------------------------------
// Types for CDNA5 WMMA (gfx1250, wave32)
// ---------------------------------------------------------------------------
typedef __bf16 bf16;
typedef bf16  v16bf __attribute__((ext_vector_type(16)));
typedef bf16  v8bf  __attribute__((ext_vector_type(8)));
typedef float v8f   __attribute__((ext_vector_type(8)));

// ---------------------------------------------------------------------------
// Problem constants (match reference)
// ---------------------------------------------------------------------------
static const int  B_   = 512;
static const int  NC_  = 64;
static const int  NE_  = 978;
static const long EC_  = 131072;
static const long EE_  = 1000000;
static const int  DIN_ = 64;
static const int  HC_  = 256;
static const int  HE_  = 128;
static const long NCn_ = (long)B_ * NC_;   // 32768 chem nodes
static const long NEn_ = (long)B_ * NE_;   // 500736 exp nodes

static inline int gblocks(long n) { return (int)((n + 255) / 256); }

// ---------------------------------------------------------------------------
// Elementwise / utility kernels
// ---------------------------------------------------------------------------
__global__ void k_zero(float* __restrict__ p, long n) {
  long i = (long)blockIdx.x * blockDim.x + threadIdx.x;
  if (i < n) p[i] = 0.0f;
}

__global__ void k_cvt_bf16(const float* __restrict__ s, bf16* __restrict__ d, long n) {
  long i = (long)blockIdx.x * blockDim.x + threadIdx.x;
  if (i < n) d[i] = (bf16)s[i];
}

// W: [K][N] f32 (row-major, as in reference x @ W) -> Wt: [N][K] bf16
__global__ void k_transpose_bf16(const float* __restrict__ W, bf16* __restrict__ Wt,
                                 int K, int N) {
  long i = (long)blockIdx.x * blockDim.x + threadIdx.x;
  if (i < (long)K * N) {
    int k = (int)(i / N);
    int n = (int)(i - (long)k * N);
    Wt[(long)n * K + k] = (bf16)W[i];
  }
}

// ---------------------------------------------------------------------------
// Graph degree / normalization
// ---------------------------------------------------------------------------
__global__ void k_deg(const int* __restrict__ col, const float* __restrict__ w,
                      float* __restrict__ deg, long E) {
  long i = (long)blockIdx.x * blockDim.x + threadIdx.x;
  if (i < E) atomicAdd(&deg[col[i]], w ? w[i] : 1.0f);
}

__global__ void k_dinv(float* __restrict__ d, long n) {
  long i = (long)blockIdx.x * blockDim.x + threadIdx.x;
  if (i < n) d[i] = rsqrtf(d[i] + 1.0f);   // +1 self loop; always > 0
}

// h[r,f] = y[r,f] * dinv[r]^2  (self-loop term) + bias[f]
__global__ void k_selfloop_bias(const float* __restrict__ y, const float* __restrict__ dinv,
                                const float* __restrict__ bias, float* __restrict__ h,
                                long rows, int H) {
  long i = (long)blockIdx.x * blockDim.x + threadIdx.x;
  if (i < rows * (long)H) {
    long r = i / H;
    int  f = (int)(i - r * H);
    float dv = dinv[r];
    h[i] = y[i] * dv * dv + bias[f];
  }
}

// per-(edge, float4-chunk) atomic scatter: h[col] += y[row] * dinv[row]*w*dinv[col]
__global__ void k_scatter(const int* __restrict__ rows, const int* __restrict__ cols,
                          const float* __restrict__ w, const float* __restrict__ dinv,
                          const float* __restrict__ y, float* __restrict__ h,
                          long E, int H) {
  const int Hq = H >> 2;
  long i = (long)blockIdx.x * blockDim.x + threadIdx.x;
  if (i >= E * (long)Hq) return;
  long e = i / Hq;
  int  q = (int)(i - e * Hq);
  int  r = rows[e];
  int  c = cols[e];
  float nw = dinv[r] * dinv[c] * (w ? w[e] : 1.0f);
  const float4 v = *(const float4*)(y + (long)r * H + q * 4);
  float* dst = h + (long)c * H + q * 4;
  atomicAdd(dst + 0, v.x * nw);
  atomicAdd(dst + 1, v.y * nw);
  atomicAdd(dst + 2, v.z * nw);
  atomicAdd(dst + 3, v.w * nw);
}

// ---------------------------------------------------------------------------
// BatchNorm (batch stats) + ELU, fused bf16 output
// ---------------------------------------------------------------------------
__global__ void k_bn_stats(const float* __restrict__ h, long rows, int H,
                           float* __restrict__ st) {       // st: [2*H] zeroed
  int f = threadIdx.x;                                     // blockDim.x == H
  long r0 = (long)blockIdx.x * 1024;
  long r1 = r0 + 1024; if (r1 > rows) r1 = rows;
  float s = 0.0f, s2 = 0.0f;
  for (long r = r0; r < r1; ++r) {
    float v = h[r * H + f];
    s += v; s2 += v * v;
  }
  atomicAdd(&st[f], s);
  atomicAdd(&st[H + f], s2);
}

__global__ void k_bn_final(const float* __restrict__ st, const float* __restrict__ g,
                           const float* __restrict__ bt, float* __restrict__ ss,
                           long rows, int H) {             // ss: [2*H] scale/shift
  int f = blockIdx.x * blockDim.x + threadIdx.x;
  if (f < H) {
    float inv = 1.0f / (float)rows;
    float mu  = st[f] * inv;
    float var = st[H + f] * inv - mu * mu;                 // biased var (== jnp.var)
    float is  = rsqrtf(var + 1e-5f);
    float sc  = g[f] * is;
    ss[f]     = sc;
    ss[H + f] = bt[f] - mu * sc;
  }
}

__global__ void k_bn_elu(const float* __restrict__ h, const float* __restrict__ ss,
                         bf16* __restrict__ xb, long rows, int H) {
  long i = (long)blockIdx.x * blockDim.x + threadIdx.x;
  if (i < rows * (long)H) {
    int f = (int)(i % H);
    float x = h[i] * ss[f] + ss[H + f];
    x = (x > 0.0f) ? x : (expf(x) - 1.0f);                 // ELU alpha=1
    xb[i] = (bf16)x;
  }
}

// mean-pool nper rows per graph, tanh, write bf16 into concatenated MLP input
__global__ void k_pool_tanh(const float* __restrict__ h, int nper, int H,
                            bf16* __restrict__ dst, int ld, long ntot) {
  long i = (long)blockIdx.x * blockDim.x + threadIdx.x;
  if (i < ntot) {
    long b = i / H;
    int  f = (int)(i - b * H);
    const float* p = h + b * (long)nper * H + f;
    float s = 0.0f;
    for (int j = 0; j < nper; ++j) s += p[(long)j * H];
    dst[b * (long)ld + f] = (bf16)tanhf(s / (float)nper);
  }
}

// ---------------------------------------------------------------------------
// WMMA bf16 GEMM: C[M][N] = A[M][K] * B[K][N] (B given pre-transposed Wt[N][K])
// Each wave owns a 16x64 output tile: one A fragment x 4 B fragments.
// Fragment layouts per CDNA5 ISA 7.12.2 (16-bit A 16x32, B 32x16, f32 C/D 16x16).
// ---------------------------------------------------------------------------
__global__ void k_wmma_gemm(const bf16* __restrict__ A, const bf16* __restrict__ Bt,
                            const float* __restrict__ bias,
                            float* __restrict__ Cf, bf16* __restrict__ Cb, int ldc,
                            int M, int K, int N, int relu) {
  const int lane = threadIdx.x & 31;
  const int wv   = threadIdx.x >> 5;
  const int tilesN = N >> 6;
  long gw  = (long)blockIdx.x * (blockDim.x >> 5) + wv;
  long tot = (long)(M >> 4) * tilesN;
  if (gw >= tot) return;                                  // wave-uniform: EXEC stays all-1
  const int tm   = (int)(gw / tilesN);
  const int tn   = (int)(gw % tilesN);
  const int half = lane >> 4;
  const int l16  = lane & 15;

  const bf16* arow = A + (long)(tm * 16 + l16) * K;
  v8f acc0 = {}, acc1 = {}, acc2 = {}, acc3 = {};

  for (int k0 = 0; k0 < K; k0 += 32) {
    // A fragment: lane(half,l16) holds row l16, K = {8h+0..7, 16+8h+0..7}
    v16bf av;
    {
      const v8bf alo = *(const v8bf*)(arow + k0 + 8 * half);
      const v8bf ahi = *(const v8bf*)(arow + k0 + 16 + 8 * half);
#pragma unroll
      for (int i = 0; i < 8; ++i) { av[i] = alo[i]; av[8 + i] = ahi[i]; }
    }
    // 4 B fragments: lane(half,l16) holds col l16, K = {8h+0..7, 16+8h+0..7}
    v8f* accs[4] = {&acc0, &acc1, &acc2, &acc3};
#pragma unroll
    for (int t = 0; t < 4; ++t) {
      const bf16* brow = Bt + (long)(tn * 64 + t * 16 + l16) * K + k0 + 8 * half;
      const v8bf blo = *(const v8bf*)(brow);
      const v8bf bhi = *(const v8bf*)(brow + 16);
      v16bf bv;
#pragma unroll
      for (int i = 0; i < 8; ++i) { bv[i] = blo[i]; bv[8 + i] = bhi[i]; }
      *accs[t] = __builtin_amdgcn_wmma_f32_16x16x32_bf16(
          false, av, false, bv, (short)0, *accs[t], false, false);
    }
  }

  // Epilogue: D element v of lane(half,l16) -> (M = 16*tm + v + 8h, N = 64*tn + 16t + l16)
  v8f accs[4] = {acc0, acc1, acc2, acc3};
#pragma unroll
  for (int t = 0; t < 4; ++t) {
    int ncol = tn * 64 + t * 16 + l16;
    float bb = bias ? bias[ncol] : 0.0f;
#pragma unroll
    for (int v = 0; v < 8; ++v) {
      int mr = tm * 16 + v + 8 * half;
      float val = accs[t][v] + bb;
      if (relu) val = fmaxf(val, 0.0f);
      long off = (long)mr * ldc + ncol;
      if (Cf) Cf[off] = val;
      if (Cb) Cb[off] = (bf16)val;
    }
  }
}

// final 128 -> 1 layer
__global__ void k_gemv(const float* __restrict__ X, const float* __restrict__ W,
                       const float* __restrict__ b, float* __restrict__ out,
                       int Bn, int K) {
  int i = blockIdx.x * blockDim.x + threadIdx.x;
  if (i < Bn) {
    float s = b[0];
    const float* x = X + (long)i * K;
    for (int k = 0; k < K; ++k) s += x[k] * W[k];
    out[i] = s;
  }
}

// ---------------------------------------------------------------------------
// Host-side helpers
// ---------------------------------------------------------------------------
static void gemm(hipStream_t s, const bf16* A, const bf16* Bt, const float* bias,
                 float* Cf, bf16* Cb, int ldc, int M, int K, int N, int relu) {
  long tot = (long)(M / 16) * (N / 64);
  int blocks = (int)((tot + 7) / 8);                       // 8 waves / 256-thread block
  k_wmma_gemm<<<blocks, 256, 0, s>>>(A, Bt, bias, Cf, Cb, ldc, M, K, N, relu);
}

struct Tower {
  const float *W[3], *b[3], *g[2], *bt[2];
};

static void run_tower(hipStream_t s, const float* X0, long rows, int nper, int din, int H,
                      const int* erow, const int* ecol, long E, const float* ew,
                      const float* dinv, const Tower& tp,
                      bf16* XB, float* Y, float* Hb, bf16* WT, float* ST,
                      bf16* poolDst, int poolLd) {
  k_cvt_bf16<<<gblocks(rows * din), 256, 0, s>>>(X0, XB, rows * din);
  for (int l = 0; l < 3; ++l) {
    int K = (l == 0) ? din : H;
    k_transpose_bf16<<<gblocks((long)K * H), 256, 0, s>>>(tp.W[l], WT, K, H);
    gemm(s, XB, WT, nullptr, Y, nullptr, H, (int)rows, K, H, 0);
    k_selfloop_bias<<<gblocks(rows * H), 256, 0, s>>>(Y, dinv, tp.b[l], Hb, rows, H);
    k_scatter<<<gblocks(E * (H / 4)), 256, 0, s>>>(erow, ecol, ew, dinv, Y, Hb, E, H);
    if (l < 2) {
      k_zero<<<1, 4 * H > 1024 ? 1024 : 4 * H, 0, s>>>(ST, 4 * H);
      k_bn_stats<<<(int)((rows + 1023) / 1024), H, 0, s>>>(Hb, rows, H, ST);
      k_bn_final<<<1, H, 0, s>>>(ST, tp.g[l], tp.bt[l], ST + 2 * H, rows, H);
      k_bn_elu<<<gblocks(rows * H), 256, 0, s>>>(Hb, ST + 2 * H, XB, rows, H);
    } else {
      k_pool_tanh<<<gblocks((long)B_ * H), 256, 0, s>>>(Hb, nper, H, poolDst, poolLd,
                                                        (long)B_ * H);
    }
  }
}

// ---------------------------------------------------------------------------
// Entry point
// ---------------------------------------------------------------------------
extern "C" void kernel_launch(void* const* d_in, const int* in_sizes, int n_in,
                              void* d_out, int out_size, void* d_ws, size_t ws_size,
                              hipStream_t stream) {
  (void)in_sizes; (void)n_in; (void)out_size; (void)ws_size;

  // ---- inputs ----
  const float* F1   = (const float*)d_in[0];   // [32768, 64]
  const float* F2   = (const float*)d_in[1];
  const int*   ADJ1 = (const int*)d_in[2];     // [2, EC]
  const int*   ADJ2 = (const int*)d_in[3];
  const float* E1   = (const float*)d_in[4];   // [500736, 64]
  const float* E2   = (const float*)d_in[5];
  const int*   EADJ = (const int*)d_in[6];     // [2, EE]
  const float* EW   = (const float*)d_in[7];   // [EE]
  // d_in[8] = syn (label; unused by forward pass)

  // ---- params (dict insertion order flatten) ----
  int p = 9;
  auto nf = [&]() { return (const float*)d_in[p++]; };
  auto loadTower = [&]() {
    Tower t;
    t.W[0] = nf(); t.W[1] = nf(); t.W[2] = nf();
    t.b[0] = nf(); t.b[1] = nf(); t.b[2] = nf();
    t.g[0] = nf(); t.bt[0] = nf(); t.g[1] = nf(); t.bt[1] = nf();
    return t;
  };
  Tower chem1 = loadTower(), chem2 = loadTower(), exp1 = loadTower(), exp2 = loadTower();
  const float *fc1W0 = nf(), *fc1W1 = nf(), *fc1b0 = nf(), *fc1b1 = nf();
  const float *fc2W0 = nf(), *fc2W1 = nf(), *fc2b0 = nf(), *fc2b1 = nf();
  const float *snW0 = nf(), *snW1 = nf(), *snW2 = nf();
  const float *snb0 = nf(), *snb1 = nf(), *snb2 = nf();

  // ---- workspace carve (256-B aligned) ----
  char* wsp = (char*)d_ws;
  auto carve = [&](size_t bytes) {
    char* q = wsp;
    wsp += (bytes + 255) & ~(size_t)255;
    return q;
  };
  float* dinvC1 = (float*)carve(NCn_ * 4);
  float* dinvC2 = (float*)carve(NCn_ * 4);
  float* dinvE  = (float*)carve(NEn_ * 4);
  bf16*  WT     = (bf16*)carve((size_t)384 * 256 * 2);     // transposed-weight scratch
  float* ST     = (float*)carve((size_t)4 * 256 * 4);      // BN sums + scale/shift
  bf16*  XB     = (bf16*)carve((size_t)NEn_ * HE_ * 2);    // activations (bf16)
  float* Y      = (float*)carve((size_t)NEn_ * HE_ * 4);   // GEMM out
  float* Hb     = (float*)carve((size_t)NEn_ * HE_ * 4);   // scatter accumulator
  bf16*  D1     = (bf16*)carve((size_t)B_ * 384 * 2);      // concat [g1c | g1e]
  bf16*  D2     = (bf16*)carve((size_t)B_ * 384 * 2);      // concat [g2c | g2e]
  bf16*  T0     = (bf16*)carve((size_t)B_ * 256 * 2);      // MLP hidden
  bf16*  XC     = (bf16*)carve((size_t)B_ * 256 * 2);      // concat [d1 | d2]
  bf16*  S0     = (bf16*)carve((size_t)B_ * 256 * 2);
  float* S1     = (float*)carve((size_t)B_ * 128 * 4);

  // ---- degree -> dinv (once per edge set; +1 self loop inside k_dinv) ----
  k_zero<<<gblocks(NCn_), 256, 0, stream>>>(dinvC1, NCn_);
  k_deg<<<gblocks(EC_), 256, 0, stream>>>(ADJ1 + EC_, nullptr, dinvC1, EC_);
  k_dinv<<<gblocks(NCn_), 256, 0, stream>>>(dinvC1, NCn_);

  k_zero<<<gblocks(NCn_), 256, 0, stream>>>(dinvC2, NCn_);
  k_deg<<<gblocks(EC_), 256, 0, stream>>>(ADJ2 + EC_, nullptr, dinvC2, EC_);
  k_dinv<<<gblocks(NCn_), 256, 0, stream>>>(dinvC2, NCn_);

  k_zero<<<gblocks(NEn_), 256, 0, stream>>>(dinvE, NEn_);
  k_deg<<<gblocks(EE_), 256, 0, stream>>>(EADJ + EE_, EW, dinvE, EE_);
  k_dinv<<<gblocks(NEn_), 256, 0, stream>>>(dinvE, NEn_);

  // ---- four GCN towers; pooled tanh outputs land directly in concat buffers ----
  run_tower(stream, F1, NCn_, NC_, DIN_, HC_, ADJ1, ADJ1 + EC_, EC_, nullptr, dinvC1,
            chem1, XB, Y, Hb, WT, ST, D1, 384);            // g1c -> D1[:, 0:256]
  run_tower(stream, F2, NCn_, NC_, DIN_, HC_, ADJ2, ADJ2 + EC_, EC_, nullptr, dinvC2,
            chem2, XB, Y, Hb, WT, ST, D2, 384);            // g2c -> D2[:, 0:256]
  run_tower(stream, E1, NEn_, NE_, DIN_, HE_, EADJ, EADJ + EE_, EE_, EW, dinvE,
            exp1, XB, Y, Hb, WT, ST, D1 + 256, 384);       // g1e -> D1[:, 256:384]
  run_tower(stream, E2, NEn_, NE_, DIN_, HE_, EADJ, EADJ + EE_, EE_, EW, dinvE,
            exp2, XB, Y, Hb, WT, ST, D2 + 256, 384);       // g2e -> D2[:, 256:384]

  // ---- fc1: [512,384] -> relu 256 -> 128 (-> XC cols 0:128) ----
  k_transpose_bf16<<<gblocks(384L * 256), 256, 0, stream>>>(fc1W0, WT, 384, 256);
  gemm(stream, D1, WT, fc1b0, nullptr, T0, 256, B_, 384, 256, 1);
  k_transpose_bf16<<<gblocks(256L * 128), 256, 0, stream>>>(fc1W1, WT, 256, 128);
  gemm(stream, T0, WT, fc1b1, nullptr, XC, 256, B_, 256, 128, 0);

  // ---- fc2 (-> XC cols 128:256) ----
  k_transpose_bf16<<<gblocks(384L * 256), 256, 0, stream>>>(fc2W0, WT, 384, 256);
  gemm(stream, D2, WT, fc2b0, nullptr, T0, 256, B_, 384, 256, 1);
  k_transpose_bf16<<<gblocks(256L * 128), 256, 0, stream>>>(fc2W1, WT, 256, 128);
  gemm(stream, T0, WT, fc2b1, nullptr, XC + 128, 256, B_, 256, 128, 0);

  // ---- snp head: 256 -> relu 256 -> relu 128 -> 1 ----
  k_transpose_bf16<<<gblocks(256L * 256), 256, 0, stream>>>(snW0, WT, 256, 256);
  gemm(stream, XC, WT, snb0, nullptr, S0, 256, B_, 256, 256, 1);
  k_transpose_bf16<<<gblocks(256L * 128), 256, 0, stream>>>(snW1, WT, 256, 128);
  gemm(stream, S0, WT, snb1, S1, nullptr, 128, B_, 256, 128, 1);
  k_gemv<<<gblocks(B_), 256, 0, stream>>>(S1, snW2, snb2, (float*)d_out, B_, 128);
}
